// SlidingWindowSelfAttention_86947317940758
// MI455X (gfx1250) — matrix operand order repaired and checked by reading, not compiled, not run on hardware
//
#include <hip/hip_runtime.h>
#include <hip/hip_bf16.h>

typedef __attribute__((ext_vector_type(16))) _Float16 v16h;
typedef __attribute__((ext_vector_type(8)))  _Float16 v8h;
typedef __attribute__((ext_vector_type(8)))  float    v8f;
typedef __attribute__((ext_vector_type(4)))  int      v4i;

#define S_LEN   4096
#define D_MODEL 1024
#define NH      16
#define HD      64
#define WIN     128
#define QS      (3 * D_MODEL)   // qkv row stride
#define BN      64              // GEMM block N tile
#define BK      32              // GEMM K step

// CDNA5 async global->LDS path (ASYNCcnt), guarded so compile never regresses.
#if defined(__AMDGCN__) && __has_builtin(__builtin_amdgcn_global_load_async_to_lds_b128) && __has_builtin(__builtin_amdgcn_s_wait_asynccnt)
#define USE_ASYNC_LDS 1
typedef __attribute__((address_space(1))) v4i g_v4i;   // global (AS1)
typedef __attribute__((address_space(3))) v4i l_v4i;   // LDS (AS3)
#else
#define USE_ASYNC_LDS 0
#endif

static __device__ __forceinline__ v16h make_v16(v8h lo, v8h hi) {
  v16h r;
#pragma unroll
  for (int i = 0; i < 8; ++i) { r[i] = lo[i]; r[i + 8] = hi[i]; }
  return r;
}

// A-fragment (M=16 x K=32, f16). Source row-major, 'stride' in elements.
// row = lane%16; lanes<16 hold K {0..7,16..23}, lanes>=16 hold K {8..15,24..31}.
static __device__ __forceinline__ v16h load_a_frag(const _Float16* base, int stride, int lane) {
  int row  = lane & 15;
  int koff = (lane & 16) ? 8 : 0;
  const _Float16* p = base + (size_t)row * stride + koff;
  return make_v16(*(const v8h*)p, *(const v8h*)(p + 16));
}

// B-fragment (K=32 x N=16, f16) from B^T rows: col = lane%16; lanes<16 hold
// K {0..7,16..23}, lanes>=16 hold K {8..15,24..31}. 'p' -> row + koff.
static __device__ __forceinline__ v16h load_b_seq(const _Float16* p) {
  return make_v16(*(const v8h*)p, *(const v8h*)(p + 16));
}

static __device__ __forceinline__ v8f wmma_f16(v16h a, v16h b, v8f c) {
  return __builtin_amdgcn_wmma_f32_16x16x32_f16(false, a, false, b, (short)0, c, false, false);
}

__global__ void cast_f32_to_f16(const float* __restrict__ in, _Float16* __restrict__ out, int n) {
  int i = blockIdx.x * blockDim.x + threadIdx.x;
  if (i < n) out[i] = (_Float16)in[i];
}

// Stage a BN x BK f16 tile of Wt (rows n0..n0+63, cols k0..k0+31) into LDS.
// 128 threads, 256 16B chunks -> 2 async (or sync) 128b ops per thread.
static __device__ __forceinline__ void stage_b(_Float16* dst, const _Float16* __restrict__ Wt,
                                               int n0, int k0, int K, int tid) {
#pragma unroll
  for (int c = tid; c < (BN * BK) / 8; c += 128) {
    int row  = c >> 2;
    int part = c & 3;
    const _Float16* g = Wt + (size_t)(n0 + row) * K + k0 + part * 8;
    _Float16* l = dst + row * BK + part * 8;
#if USE_ASYNC_LDS
    __builtin_amdgcn_global_load_async_to_lds_b128((g_v4i*)g, (l_v4i*)l, 0, 0);
#else
    *(v8h*)l = *(const v8h*)g;
#endif
  }
}

// C[M,N] = A[M,K] @ Wt[N,K]^T + bias; optional scale on first 'qcols' columns;
// f16 or f32 output. One wave computes 32x64; 4 waves cover 128x64 and share
// the B tile through LDS (double-buffered async copy).
__global__ void gemm_wmma(const _Float16* __restrict__ A,
                          const _Float16* __restrict__ Wt,
                          const float* __restrict__ bias,
                          void* __restrict__ out,
                          int M, int N, int K,
                          int out_f16, float qscale, int qcols) {
  __shared__ _Float16 bsm[2][BN][BK];   // 2 x 4KB

  int tid  = threadIdx.x;
  int lane = tid & 31;
  int wave = tid >> 5;
  int n0 = blockIdx.x * BN;
  int m0 = blockIdx.y * 128 + wave * 32;
  int row15 = lane & 15;
  int klo   = (lane & 16) ? 16 : 0;

  v8f z;
#pragma unroll
  for (int i = 0; i < 8; ++i) z[i] = 0.f;
  v8f c[2][4];
#pragma unroll
  for (int i = 0; i < 2; ++i)
#pragma unroll
    for (int j = 0; j < 4; ++j) c[i][j] = z;

  stage_b(&bsm[0][0][0], Wt, n0, 0, K, tid);
  int buf = 0;

  for (int k0 = 0; k0 < K; k0 += BK) {
    if (k0 + BK < K) {
      stage_b(&bsm[buf ^ 1][0][0], Wt, n0, k0 + BK, K, tid);
#if USE_ASYNC_LDS
      __builtin_amdgcn_s_wait_asynccnt(2);   // retire the 'buf' group
#endif
    } else {
#if USE_ASYNC_LDS
      __builtin_amdgcn_s_wait_asynccnt(0);
#endif
    }
    __syncthreads();                         // buf tile visible to all waves

    v16h a0 = load_a_frag(A + (size_t)m0 * K + k0, K, lane);
    v16h a1 = load_a_frag(A + (size_t)(m0 + 16) * K + k0, K, lane);
#pragma unroll
    for (int j = 0; j < 4; ++j) {
      v16h b = load_b_seq(&bsm[buf][j * 16 + row15][klo >> 1]);
      c[0][j] = wmma_f16(a0, b, c[0][j]);
      c[1][j] = wmma_f16(a1, b, c[1][j]);
    }
    __syncthreads();                         // everyone done reading buf
    buf ^= 1;
  }

  int rbase = (lane & 16) ? 8 : 0;
#pragma unroll
  for (int i = 0; i < 2; ++i)
#pragma unroll
    for (int j = 0; j < 4; ++j) {
      int col = n0 + j * 16 + row15;
      float b = bias[col];
      float s = (col < qcols) ? qscale : 1.0f;
#pragma unroll
      for (int r = 0; r < 8; ++r) {
        int row = m0 + i * 16 + rbase + r;
        float v = (c[i][j][r] + b) * s;
        if (out_f16) ((_Float16*)out)[(size_t)row * N + col] = (_Float16)v;
        else         ((float*)out)[(size_t)row * N + col] = v;
      }
    }
}

// Flash attention over a +/-128 sliding window. One wave per (head, 16-query
// tile). Fixed 9 key-steps of 32 per tile (uniform across waves -> barriers
// safe). Q in qkv is pre-scaled by 1/sqrt(hd).
__global__ void attn_wmma(const _Float16* __restrict__ qkv, _Float16* __restrict__ outh) {
  __shared__ float    sc[4][16][32];
  __shared__ _Float16 pl[4][16][32];
  __shared__ float    scl[4][16];
  __shared__ float    lsum[4][16];

  int lane = threadIdx.x & 31;
  int wave = threadIdx.x >> 5;
  int tile = blockIdx.x * 4 + wave;
  int h  = tile >> 8;           // / (S/16)
  int q0 = (tile & 255) * 16;

  const _Float16* qbase = qkv + h * HD;
  const _Float16* kbase = qkv + D_MODEL + h * HD;
  const _Float16* vbase = qkv + 2 * D_MODEL + h * HD;

  int row15 = lane & 15;
  int rbase = (lane & 16) ? 8 : 0;
  int klo   = (lane & 16) ? 16 : 0;
  int koffA = (lane & 16) ? 8 : 0;

  v16h qa0 = load_a_frag(qbase + (size_t)q0 * QS,      QS, lane);
  v16h qa1 = load_a_frag(qbase + (size_t)q0 * QS + 32, QS, lane);

  v8f zero;
#pragma unroll
  for (int i = 0; i < 8; ++i) zero[i] = 0.f;
  v8f oc[4];
#pragma unroll
  for (int j = 0; j < 4; ++j) oc[j] = zero;

  float m_r = -1e30f, l_r = 0.f;   // live only in lanes 0..15

  int kb0 = q0 - 144;              // covers keys q0-128 .. q0+143 in 9 steps
  for (int step = 0; step < 9; ++step) {
    int kb = kb0 + step * 32;

    // ---- scores: two 16x16 tiles (32 keys), Kdim = 64 channels ----
#pragma unroll
    for (int n = 0; n < 2; ++n) {
      int t  = kb + n * 16 + row15;
      int tc = t < 0 ? 0 : (t >= S_LEN ? S_LEN - 1 : t);
      const _Float16* p = kbase + (size_t)tc * QS + koffA;
      v16h kf0 = make_v16(*(const v8h*)p,        *(const v8h*)(p + 16));
      v16h kf1 = make_v16(*(const v8h*)(p + 32), *(const v8h*)(p + 48));
      v8f s = wmma_f16(qa0, kf0, zero);
      s = wmma_f16(qa1, kf1, s);
#pragma unroll
      for (int r = 0; r < 8; ++r) sc[wave][rbase + r][n * 16 + row15] = s[r];
    }
    __syncthreads();

    // ---- online softmax: lane r owns row r ----
    if (lane < 16) {
      int r = lane, qrow = q0 + r;
      float mx = m_r;
      float sv[32];
#pragma unroll
      for (int cc = 0; cc < 32; ++cc) {
        int t = kb + cc;
        bool ok = (t >= 0) && (t < S_LEN) && (t >= qrow - WIN) && (t <= qrow + WIN);
        float v = ok ? sc[wave][r][cc] : -1e30f;
        sv[cc] = v;
        mx = fmaxf(mx, v);
      }
      float rescale = __expf(m_r - mx);
      float ps = 0.f;
#pragma unroll
      for (int cc = 0; cc < 32; ++cc) {
        float p = (sv[cc] > -1e29f) ? __expf(sv[cc] - mx) : 0.f;
        ps += p;
        pl[wave][r][cc] = (_Float16)p;
      }
      l_r = l_r * rescale + ps;
      m_r = mx;
      scl[wave][r] = rescale;
    }
    __syncthreads();

    // ---- rescale O, then O += P @ V ----
    float rs[8];
#pragma unroll
    for (int r = 0; r < 8; ++r) rs[r] = scl[wave][rbase + r];
#pragma unroll
    for (int j = 0; j < 4; ++j)
#pragma unroll
      for (int r = 0; r < 8; ++r) oc[j][r] *= rs[r];

    const _Float16* pp = &pl[wave][row15][koffA];
    v16h pa = make_v16(*(const v8h*)pp, *(const v8h*)(pp + 16));

#pragma unroll
    for (int j = 0; j < 4; ++j) {
      int cch = j * 16 + row15;
      v16h vf;
#pragma unroll
      for (int i = 0; i < 16; ++i) {
        int t  = kb + koffA + (i & 7) + ((i & 8) << 1);
        int tc = t < 0 ? 0 : (t >= S_LEN ? S_LEN - 1 : t);
        vf[i] = vbase[(size_t)tc * QS + cch];
      }
      oc[j] = wmma_f16(pa, vf, oc[j]);
    }
    __syncthreads();
  }

  if (lane < 16) lsum[wave][lane] = l_r;
  __syncthreads();

  float li[8];
#pragma unroll
  for (int r = 0; r < 8; ++r) li[r] = 1.0f / lsum[wave][rbase + r];
#pragma unroll
  for (int j = 0; j < 4; ++j)
#pragma unroll
    for (int r = 0; r < 8; ++r) {
      int row = q0 + rbase + r;
      int col = h * HD + j * 16 + row15;
      outh[(size_t)row * D_MODEL + col] = (_Float16)(oc[j][r] * li[r]);
    }
}

extern "C" void kernel_launch(void* const* d_in, const int* in_sizes, int n_in,
                              void* d_out, int out_size, void* d_ws, size_t ws_size,
                              hipStream_t stream) {
  const float* x     = (const float*)d_in[0];   // [4096,1,1024]
  const float* w_in  = (const float*)d_in[1];   // [3072,1024]
  const float* b_in  = (const float*)d_in[2];   // [3072]
  const float* w_out = (const float*)d_in[3];   // [1024,1024]
  const float* b_out = (const float*)d_in[4];   // [1024]
  float* y = (float*)d_out;                     // [4096,1,1024]

  char* ws = (char*)d_ws;
  size_t off = 0;
  _Float16* xh    = (_Float16*)(ws + off); off += (size_t)S_LEN * D_MODEL * 2;
  _Float16* wInH  = (_Float16*)(ws + off); off += (size_t)3 * D_MODEL * D_MODEL * 2;
  _Float16* wOutH = (_Float16*)(ws + off); off += (size_t)D_MODEL * D_MODEL * 2;
  _Float16* qkvh  = (_Float16*)(ws + off); off += (size_t)S_LEN * 3 * D_MODEL * 2;
  _Float16* attnH = (_Float16*)(ws + off); off += (size_t)S_LEN * D_MODEL * 2;

  int n;
  n = S_LEN * D_MODEL;
  cast_f32_to_f16<<<(n + 255) / 256, 256, 0, stream>>>(x, xh, n);
  n = 3 * D_MODEL * D_MODEL;
  cast_f32_to_f16<<<(n + 255) / 256, 256, 0, stream>>>(w_in, wInH, n);
  n = D_MODEL * D_MODEL;
  cast_f32_to_f16<<<(n + 255) / 256, 256, 0, stream>>>(w_out, wOutH, n);

  // QKV projection; fold 1/sqrt(hd) into Q columns, emit f16.
  dim3 g1(3 * D_MODEL / BN, S_LEN / 128);
  gemm_wmma<<<g1, 128, 0, stream>>>(xh, wInH, b_in, qkvh,
                                    S_LEN, 3 * D_MODEL, D_MODEL,
                                    /*out_f16=*/1, /*qscale=*/0.125f, /*qcols=*/D_MODEL);

  // Sliding-window flash attention: 16 heads * 256 query tiles / 4 waves.
  attn_wmma<<<(NH * (S_LEN / 16)) / 4, 128, 0, stream>>>(qkvh, attnH);

  // Output projection to fp32.
  dim3 g2(D_MODEL / BN, S_LEN / 128);
  gemm_wmma<<<g2, 128, 0, stream>>>(attnH, wOutH, b_out, y,
                                    S_LEN, D_MODEL, D_MODEL,
                                    /*out_f16=*/0, /*qscale=*/1.0f, /*qcols=*/0);
}
